// CustomLSTM_3246995276002
// MI455X (gfx1250) — matrix operand order repaired and checked by reading, not compiled, not run on hardware
//
#include <hip/hip_runtime.h>
#include <hip/hip_bf16.h>
#include <stdint.h>

// ---------------------------------------------------------------------------
// CDNA5 / gfx1250 LSTM. wave32, WMMA bf16 16x16x32, fp32 accumulate.
// Persistent recurrent scan: U cached in LDS, grid sync via L2 atomics.
// ---------------------------------------------------------------------------

typedef __attribute__((ext_vector_type(16))) __bf16 v16bf;
typedef __attribute__((ext_vector_type(8)))  float  v8f;

#define HS    512
#define BATCH 128
#define SEQ   512
#define INSZ  96
#define G4    2048   // 4*HS
#define NWG   32     // workgroups in the persistent scan (16 j-cols each)

// ---- workspace layout (bytes, all 256-aligned) ----
#define OFF_XBF    ((size_t)0)            // x as bf16        [65536][96]
#define OFF_WPP    ((size_t)12582912)     // Wp packed B-frag  (32 ntiles x 3 ksteps)
#define OFF_XPROJ  ((size_t)12681216)     // x_proj bf16      [65536][512]
#define OFF_WP     ((size_t)79790080)     // W packed B-frag   (128 x 16)
#define OFF_UP     ((size_t)81887232)     // U packed B-frag   (128 x 16)
#define OFF_FCWP   ((size_t)83984384)     // fc_W packed B-frag (6 x 16)
#define OFF_XW     ((size_t)84082688)     // xW bf16          [512][2048][128]
#define OFF_HBF    ((size_t)352518144)    // h bf16, 2 buffers [2][128][512]
#define OFF_BAR    ((size_t)352780288)    // grid-sync counter (1 u32)

// ---------------- helpers ----------------
__device__ __forceinline__ unsigned short f2bf(float f) {
    union { float f; uint32_t u; } v; v.f = f;
    uint32_t r = v.u + 0x7FFFu + ((v.u >> 16) & 1u);   // RNE
    return (unsigned short)(r >> 16);
}
__device__ __forceinline__ float bf2f(uint32_t h) {
    union { uint32_t u; float f; } v; v.u = h << 16; return v.f;
}
__device__ __forceinline__ float sigf(float x) {
    return 1.0f / (1.0f + __expf(-x));
}

union AFrag { v16bf v; uint4 q[2]; };
union BFrag { v16bf v; uint4 q[2]; };
union CAcc  { v8f   v; float f[8]; };

// Load 16x32 bf16 A-fragment from row-major bf16 (ld elements).
__device__ __forceinline__ AFrag load_a(const unsigned short* __restrict__ src,
                                        int ld, int row, int k0, int ka) {
    AFrag a;
    const unsigned short* p = src + (size_t)row * ld + k0 + ka;
    a.q[0] = *(const uint4*)(p);
    a.q[1] = *(const uint4*)(p + 16);
    return a;
}

// Load pre-packed B-fragment block from global (32 lanes x 16 bf16, contiguous).
__device__ __forceinline__ BFrag load_b(const unsigned short* __restrict__ packed,
                                        int block, int lane) {
    BFrag b;
    const unsigned short* p = packed + ((size_t)block * 32 + lane) * 16;
    b.q[0] = *(const uint4*)(p);
    b.q[1] = *(const uint4*)(p + 8);
    return b;
}

// ---------------- weight packing: fp32 [K][N] row-major -> B-fragment blocks ----
__global__ void pack_b_kernel(const float* __restrict__ src,
                              unsigned short* __restrict__ dst,
                              int N, int ksteps, int nblocks) {
    int idx = blockIdx.x * blockDim.x + threadIdx.x;
    if (idx >= nblocks * 32) return;
    int lane  = idx & 31;
    int block = idx >> 5;
    int ntile = block / ksteps;
    int kstep = block - ntile * ksteps;
    int k0 = kstep * 32 + ((lane >> 4) << 3);
    int n  = ntile * 16 + (lane & 15);
    unsigned short* d = dst + (size_t)idx * 16;
#pragma unroll
    for (int e = 0; e < 16; ++e)
        d[e] = f2bf(src[(size_t)(k0 + e + ((e >> 3) << 3)) * N + n]);
}

// ---------------- fp32 -> bf16 ----------------
__global__ void cvt_bf16_kernel(const float* __restrict__ src,
                                unsigned short* __restrict__ dst, int n) {
    int i = blockIdx.x * blockDim.x + threadIdx.x;
    if (i < n) dst[i] = f2bf(src[i]);
}

// ---------------- init: zero h (both buffers), c, and the sync counter --------
__global__ void init_state_kernel(unsigned short* __restrict__ hbf,
                                  float* __restrict__ cF,
                                  unsigned int* __restrict__ bar) {
    int i = blockIdx.x * blockDim.x + threadIdx.x;
    if (i == 0) *bar = 0u;
    if (i < BATCH * HS) {
        cF[i] = 0.0f;
        hbf[i] = 0;
        hbf[i + BATCH * HS] = 0;
    }
}

// ---------------- GEMM1: x_proj = x_bf16[65536x96] @ Wp + bp -> bf16[65536x512]
__global__ __launch_bounds__(256) void gemm1_kernel(
        const unsigned short* __restrict__ xb,
        const unsigned short* __restrict__ wpp,
        const float* __restrict__ bp,
        unsigned short* __restrict__ xproj) {
    const int lane = threadIdx.x & 31, wave = threadIdx.x >> 5;
    const int lhalf = lane >> 4, lmod = lane & 15;
    const int M0 = blockIdx.x * 128, N0 = blockIdx.y * 64;
    const int m0 = M0 + wave * 16;
    const int ka = lhalf * 8;
    CAcc acc[4];
#pragma unroll
    for (int ct = 0; ct < 4; ++ct) {
        float b = bp[N0 + ct * 16 + lmod];
#pragma unroll
        for (int r = 0; r < 8; ++r) acc[ct].f[r] = b;
    }
#pragma unroll
    for (int ks = 0; ks < 3; ++ks) {
        AFrag a = load_a(xb, INSZ, m0 + lmod, ks * 32, ka);
#pragma unroll
        for (int ct = 0; ct < 4; ++ct) {
            BFrag b = load_b(wpp, ((N0 >> 4) + ct) * 3 + ks, lane);
            acc[ct].v = __builtin_amdgcn_wmma_f32_16x16x32_bf16(
                false, a.v, false, b.v, (short)0, acc[ct].v, false, false);
        }
    }
#pragma unroll
    for (int ct = 0; ct < 4; ++ct)
#pragma unroll
        for (int r = 0; r < 8; ++r) {
            int row = m0 + r + 8 * lhalf;
            xproj[(size_t)row * HS + N0 + ct * 16 + lmod] = f2bf(acc[ct].f[r]);
        }
}

// ---------------- GEMM2: xW[s][n][b] = x_proj @ W + bias  (bf16 out) ----------
__global__ __launch_bounds__(256) void gemm2_kernel(
        const unsigned short* __restrict__ xproj,
        const unsigned short* __restrict__ wp,
        const float* __restrict__ bias,
        unsigned short* __restrict__ xw) {
    const int lane = threadIdx.x & 31, wave = threadIdx.x >> 5;
    const int lhalf = lane >> 4, lmod = lane & 15;
    const int M0 = blockIdx.x * 128, N0 = blockIdx.y * 64;
    const int m0 = M0 + wave * 16;
    const int ka = lhalf * 8;
    CAcc acc[4];
#pragma unroll
    for (int ct = 0; ct < 4; ++ct) {
        float b = bias[N0 + ct * 16 + lmod];
#pragma unroll
        for (int r = 0; r < 8; ++r) acc[ct].f[r] = b;
    }
    for (int ks = 0; ks < 16; ++ks) {
        AFrag a = load_a(xproj, HS, m0 + lmod, ks * 32, ka);
        if (ks < 15)
            __builtin_prefetch(xproj + (size_t)(m0 + lmod) * HS + (ks + 1) * 32, 0, 0);
#pragma unroll
        for (int ct = 0; ct < 4; ++ct) {
            BFrag b = load_b(wp, ((N0 >> 4) + ct) * 16 + ks, lane);
            acc[ct].v = __builtin_amdgcn_wmma_f32_16x16x32_bf16(
                false, a.v, false, b.v, (short)0, acc[ct].v, false, false);
        }
    }
#pragma unroll
    for (int ct = 0; ct < 4; ++ct)
#pragma unroll
        for (int r = 0; r < 8; ++r) {
            int g = m0 + r + 8 * lhalf;
            int bidx = g >> 9, s = g & 511;
            int n = N0 + ct * 16 + lmod;
            xw[((size_t)s * G4 + n) * BATCH + bidx] = f2bf(acc[ct].f[r]);
        }
}

// ---------------- persistent recurrent scan ------------------------------------
// grid = NWG wgs (16 j-columns each), 8 waves (16 batch rows each).
// Each wg stages its 64KB slice of packed U into LDS once, then loops all 512
// timesteps, syncing the grid with a monotonic L2 atomic counter per step.
// LDS block layout (per gate g, kstep ks): q0[lane] at lane*16B, q1[lane] at
// 512B + lane*16B  -> conflict-free ds_load_b128.
__global__ __launch_bounds__(256) void lstm_scan_kernel(
        const unsigned short* __restrict__ xw,
        const unsigned short* __restrict__ up,
        unsigned short* __restrict__ hbf,
        float* __restrict__ cF,
        float* __restrict__ hF,
        unsigned int* __restrict__ bar) {
    const int lane = threadIdx.x & 31, wave = threadIdx.x >> 5;
    const int lhalf = lane >> 4, lmod = lane & 15;
    const int j0 = blockIdx.x * 16;
    const int m0 = wave * 16;
    const int ka = lhalf * 8;

    extern __shared__ uint4 ldsq[];   // 4096 x uint4 = 64KB: [4 gates][16 ksteps][64]

    // ---- stage this workgroup's 4 U column-tiles (4 x 16KB) into LDS ----
    const uint4* upq = (const uint4*)up;
#pragma unroll
    for (int it = 0; it < 16; ++it) {
        int idx = it * 256 + threadIdx.x;        // 0..4095
        int blk = idx >> 6;                      // local block: g*16+ks
        int u   = idx & 63;                      // uint4 within block
        int l   = u >> 1, qi = u & 1;
        int g   = blk >> 4, ks = blk & 15;
        int ntile = g * 32 + (j0 >> 4);
        ldsq[blk * 64 + qi * 32 + l] = upq[(size_t)(ntile * 16 + ks) * 64 + u];
    }
    __syncthreads();

    unsigned int expected = 0;
    for (int t = 0; t < SEQ; ++t) {
        const unsigned short* h_rd = hbf + (size_t)(t & 1) * (BATCH * HS);
        unsigned short*       h_wr = hbf + (size_t)((t + 1) & 1) * (BATCH * HS);

        // C init = xW[t] fragment (one b128 load per gate tile)
        CAcc acc[4];
#pragma unroll
        for (int g = 0; g < 4; ++g) {
            int n = g * HS + j0 + lmod;
            const unsigned short* p =
                xw + ((size_t)t * G4 + n) * BATCH + m0 + 8 * lhalf;
            uint4 q = *(const uint4*)p;
            acc[g].f[0] = bf2f(q.x & 0xffff); acc[g].f[1] = bf2f(q.x >> 16);
            acc[g].f[2] = bf2f(q.y & 0xffff); acc[g].f[3] = bf2f(q.y >> 16);
            acc[g].f[4] = bf2f(q.z & 0xffff); acc[g].f[5] = bf2f(q.z >> 16);
            acc[g].f[6] = bf2f(q.w & 0xffff); acc[g].f[7] = bf2f(q.w >> 16);
        }

        // prefetch next step's xW slab while we compute this one
        if (t < SEQ - 1) {
#pragma unroll
            for (int g = 0; g < 4; ++g) {
                int n = g * HS + j0 + lmod;
                __builtin_prefetch(
                    xw + ((size_t)(t + 1) * G4 + n) * BATCH + m0 + 8 * lhalf, 0, 0);
            }
        }

        // gates += h @ U   (A from global/L2, B from LDS)
        for (int ks = 0; ks < 16; ++ks) {
            AFrag a = load_a(h_rd, HS, m0 + lmod, ks * 32, ka);
#pragma unroll
            for (int g = 0; g < 4; ++g) {
                BFrag b;
                b.q[0] = ldsq[(g * 16 + ks) * 64 + lane];
                b.q[1] = ldsq[(g * 16 + ks) * 64 + 32 + lane];
                acc[g].v = __builtin_amdgcn_wmma_f32_16x16x32_bf16(
                    false, a.v, false, b.v, (short)0, acc[g].v, false, false);
            }
        }

        // elementwise LSTM update (each (b,j) owned by exactly one lane)
#pragma unroll
        for (int r = 0; r < 8; ++r) {
            int b = m0 + r + 8 * lhalf;
            int j = j0 + lmod;
            float ig = sigf(acc[0].f[r]);
            float fg = sigf(acc[1].f[r]);
            float gg = tanhf(acc[2].f[r]);
            float og = sigf(acc[3].f[r]);
            size_t ix = (size_t)b * HS + j;
            float c = fg * cF[ix] + ig * gg;
            float h = og * tanhf(c);
            cF[ix] = c;
            hF[ix] = h;
            h_wr[ix] = f2bf(h);
        }

        // ---- grid-wide barrier: all h writes of step t before any reads of t+1
        __threadfence();
        __syncthreads();
        if (threadIdx.x == 0)
            __hip_atomic_fetch_add(bar, 1u, __ATOMIC_RELEASE,
                                   __HIP_MEMORY_SCOPE_AGENT);
        expected += NWG;
        if (threadIdx.x == 0) {
            while (__hip_atomic_load(bar, __ATOMIC_ACQUIRE,
                                     __HIP_MEMORY_SCOPE_AGENT) < expected)
                __builtin_amdgcn_s_sleep(1);
        }
        __syncthreads();
    }
}

// ---------------- final FC: pred = h @ fc_W + fc_b  (M=128,N=96,K=512) --------
__global__ __launch_bounds__(256) void pred_kernel(
        const unsigned short* __restrict__ hbf,
        const unsigned short* __restrict__ fcp,
        const float* __restrict__ fcb,
        float* __restrict__ out) {
    const int lane = threadIdx.x & 31, wave = threadIdx.x >> 5;
    const int lhalf = lane >> 4, lmod = lane & 15;
    const int m0 = wave * 16;
    const int ka = lhalf * 8;
    CAcc acc[6];
#pragma unroll
    for (int ct = 0; ct < 6; ++ct) {
        float b = fcb[ct * 16 + lmod];
#pragma unroll
        for (int r = 0; r < 8; ++r) acc[ct].f[r] = b;
    }
    for (int ks = 0; ks < 16; ++ks) {
        AFrag a = load_a(hbf, HS, m0 + lmod, ks * 32, ka);
#pragma unroll
        for (int ct = 0; ct < 6; ++ct) {
            BFrag b = load_b(fcp, ct * 16 + ks, lane);
            acc[ct].v = __builtin_amdgcn_wmma_f32_16x16x32_bf16(
                false, a.v, false, b.v, (short)0, acc[ct].v, false, false);
        }
    }
#pragma unroll
    for (int ct = 0; ct < 6; ++ct)
#pragma unroll
        for (int r = 0; r < 8; ++r) {
            int b = m0 + r + 8 * lhalf;
            out[(size_t)b * INSZ + ct * 16 + lmod] = acc[ct].f[r];
        }
}

// ---------------------------------------------------------------------------
extern "C" void kernel_launch(void* const* d_in, const int* in_sizes, int n_in,
                              void* d_out, int out_size, void* d_ws, size_t ws_size,
                              hipStream_t stream) {
    const float* x    = (const float*)d_in[0];   // [128,512,96]
    const float* Wp   = (const float*)d_in[1];   // [96,512]
    const float* bp   = (const float*)d_in[2];   // [512]
    const float* W    = (const float*)d_in[3];   // [512,2048]
    const float* U    = (const float*)d_in[4];   // [512,2048]
    const float* bias = (const float*)d_in[5];   // [2048]
    const float* fcW  = (const float*)d_in[6];   // [512,96]
    const float* fcb  = (const float*)d_in[7];   // [96]

    char* ws = (char*)d_ws;
    unsigned short* xbf   = (unsigned short*)(ws + OFF_XBF);
    unsigned short* wpp   = (unsigned short*)(ws + OFF_WPP);
    unsigned short* xproj = (unsigned short*)(ws + OFF_XPROJ);
    unsigned short* wP    = (unsigned short*)(ws + OFF_WP);
    unsigned short* uP    = (unsigned short*)(ws + OFF_UP);
    unsigned short* fcP   = (unsigned short*)(ws + OFF_FCWP);
    unsigned short* xw    = (unsigned short*)(ws + OFF_XW);
    unsigned short* hbf   = (unsigned short*)(ws + OFF_HBF);
    unsigned int*   bar   = (unsigned int*)(ws + OFF_BAR);

    float* out  = (float*)d_out;
    float* pred = out;                       // [128,96]
    float* hF   = out + BATCH * INSZ;        // [128,512]
    float* cF   = hF + BATCH * HS;           // [128,512]

    // 1) conversions & weight packing
    {
        int n = BATCH * SEQ * INSZ;
        cvt_bf16_kernel<<<(n + 255) / 256, 256, 0, stream>>>(x, xbf, n);
    }
    pack_b_kernel<<<(32 * 3 * 32 + 255) / 256, 256, 0, stream>>>(Wp, wpp, HS, 3, 32 * 3);
    pack_b_kernel<<<(128 * 16 * 32 + 255) / 256, 256, 0, stream>>>(W, wP, G4, 16, 128 * 16);
    pack_b_kernel<<<(128 * 16 * 32 + 255) / 256, 256, 0, stream>>>(U, uP, G4, 16, 128 * 16);
    pack_b_kernel<<<(6 * 16 * 32 + 255) / 256, 256, 0, stream>>>(fcW, fcP, INSZ, 16, 6 * 16);
    init_state_kernel<<<(BATCH * HS + 255) / 256, 256, 0, stream>>>(hbf, cF, bar);

    // 2) batched projection GEMMs
    gemm1_kernel<<<dim3(512, 8), 256, 0, stream>>>(xbf, wpp, bp, xproj);
    gemm2_kernel<<<dim3(512, 32), 256, 0, stream>>>(xproj, wP, bias, xw);

    // 3) persistent LSTM scan: one launch, 512 internal steps, 64KB LDS per wg
    lstm_scan_kernel<<<NWG, 256, 65536, stream>>>(xw, uP, hbf, cF, hF, bar);

    // 4) final FC (h after step 511 lives in buffer 0 since 512 is even)
    pred_kernel<<<1, 256, 0, stream>>>(hbf, fcP, fcb, pred);
}
